// Seq2SeqModel_69114613729154
// MI455X (gfx1250) — compile-verified, matrix operand order checked by reading, not run on hardware
//
#include <hip/hip_runtime.h>

// ---------------------------------------------------------------------------
// Seq2Seq LSTM (H=2048, N=4096) persistent-kernel implementation for MI455X.
//
// Key decisions (reasoned from MI455X specs, compile-only):
//  * Recurrence => no time batching. Cost/step = W_hh@h (8192x2048 matvec).
//  * fp32 streaming of W_hh from HBM would be 512 GB total (~22ms @23.3TB/s).
//    Instead: bf16 weights, pinned in LDS. 128 WGs x 256KB LDS slices hold an
//    entire 32MB bf16 W_hh; per-step weight reads hit LDS, not HBM/L2.
//  * WMMA bf16 (v_wmma_f32_16x16x32_bf16), A = h replicated over M rows,
//    B = W^T tile (pre-swizzled into the exact lane layout by a prep kernel),
//    so D[m][n] = y_n replicated: lanes 0..15 of acc VGPR0 hold the outputs.
//  * Per-step h staging and per-phase weight fill use the CDNA5 async path:
//    global_load_async_to_lds_b128 + s_wait_asynccnt (no VGPR round-trip;
//    the copy sits between two grid barriers so its latency is fully exposed
//    8192 times — the async engine shortens that critical path).
//  * Grid-wide step sync: ticket barrier (atomicAdd + agent-scope load +
//    s_sleep). 2*(4096+1) rounds, counter < 2^21 * 128 — no overflow.
// ---------------------------------------------------------------------------

#define HDIM 2048
#define SEQ  4096
#define NWG  128          // 2048 / 16 hidden units per workgroup
#define BLK  256          // 8 wave32
#define SLICE 16          // hidden units owned per WG
#define KCHUNKS 64        // 2048 / 32 (K per WMMA)
#define FRAG_ELEMS (4 * KCHUNKS * 32 * 16)   // bf16 elements per WG = 131072

// LDS layout (byte offsets into dynamic shared)
#define WL_OFF   0u        // 262144 B : weight fragments
#define HL_OFF   262144u   //   4096 B : staged h vector
#define PART_OFF 266240u   //   2048 B : per-wave gate partials

typedef __attribute__((ext_vector_type(16))) __bf16 v16bf;
typedef __attribute__((ext_vector_type(8)))  float  v8f;

__device__ __forceinline__ float sigf(float x) {
    return 1.0f / (1.0f + __expf(-x));
}

// Async copy of one 16B chunk per lane: global -> LDS via the ASYNCcnt path.
// ldsOff is the byte offset within this workgroup's LDS allocation (HW adds
// the wave's LDS_BASE); gaddr is the 64-bit global source address.
__device__ __forceinline__ void asyncLds128(unsigned ldsOff, unsigned long long gaddr) {
    asm volatile("global_load_async_to_lds_b128 %0, %1, off"
                 :: "v"(ldsOff), "v"(gaddr)
                 : "memory");
}
__device__ __forceinline__ void waitAsync0() {
    asm volatile("s_wait_asynccnt 0x0" ::: "memory");
}

// Device-wide ticket barrier: round r completes when counter reaches r*NWG.
__device__ __forceinline__ void gridBarrier(unsigned* cnt, unsigned round) {
    __syncthreads();
    __threadfence();                       // publish this WG's global writes
    if (threadIdx.x == 0) {
        atomicAdd(cnt, 1u);
        const unsigned target = round * NWG;
        while (__hip_atomic_load(cnt, __ATOMIC_ACQUIRE,
                                 __HIP_MEMORY_SCOPE_AGENT) < target) {
            __builtin_amdgcn_s_sleep(2);
        }
    }
    __syncthreads();
}

// ---------------------------------------------------------------------------
// Prep: swizzle fp32 W_hh (4H x H, row major) into bf16 WMMA B-operand
// fragments, grouped per WG / gate block / K-chunk:
//   out[(((g*4+q)*64+kc)*32+lane)*16+e] = bf16( W[q*H+g*16+n][kc*32+k] )
// with n = lane&15, hi = lane>>4, r = e>>1, s = e&1,
//      k = (r&4?16:0) + hi*8 + (r&3)*2 + s   (mirrors the 16-bit A/B layout).
// ---------------------------------------------------------------------------
__global__ void swizzle_whh(const float* __restrict__ W, __bf16* __restrict__ out) {
    const unsigned u  = blockIdx.x * blockDim.x + threadIdx.x;   // < 2^24
    const unsigned e  = u & 15u;
    const unsigned ln = (u >> 4) & 31u;
    const unsigned kc = (u >> 9) & 63u;
    const unsigned q  = (u >> 15) & 3u;
    const unsigned g  = u >> 17;
    const unsigned n  = ln & 15u, hi = ln >> 4;
    const unsigned r  = e >> 1,   s  = e & 1u;
    const unsigned k  = ((r & 4u) ? 16u : 0u) + hi * 8u + ((r & 3u) << 1) + s;
    const unsigned row = q * HDIM + g * SLICE + n;
    const unsigned col = kc * 32u + k;
    out[u] = (__bf16)W[(size_t)row * HDIM + col];
}

// Prep: fused biases, zero h0, reset barrier counter (re-run every launch so
// graph replay stays deterministic).
__global__ void init_kernel(const float* __restrict__ ebi, const float* __restrict__ ebh,
                            const float* __restrict__ dbi, const float* __restrict__ dbh,
                            float* __restrict__ biasE, float* __restrict__ biasD,
                            __bf16* __restrict__ hbuf, unsigned* __restrict__ cnt) {
    const unsigned i = blockIdx.x * blockDim.x + threadIdx.x;
    if (i == 0) *cnt = 0u;
    if (i < 4 * HDIM) {
        biasE[i] = ebi[i] + ebh[i];
        biasD[i] = dbi[i] + dbh[i];
    }
    if (i < HDIM) hbuf[i] = (__bf16)0.0f;
}

// Finalize: y[t] = out_b + sum_g ypart[t][g]
__global__ void finalize_kernel(const float* __restrict__ ypart,
                                const float* __restrict__ out_b,
                                float* __restrict__ y) {
    const unsigned t = blockIdx.x * blockDim.x + threadIdx.x;
    if (t < SEQ) {
        float s = out_b[0];
        const float* p = ypart + (size_t)t * NWG;
#pragma unroll 8
        for (int g = 0; g < NWG; ++g) s += p[g];
        y[t] = s;
    }
}

// ---------------------------------------------------------------------------
// Persistent recurrent kernel: 128 WGs, each owns 16 hidden units.
// LDS (dynamic, 263 KB): [ Wfrag 256KB | h 4KB | part 2KB | gate/win/bias/y/c ]
// ---------------------------------------------------------------------------
__global__ void __launch_bounds__(BLK)
lstm_persistent(const float* __restrict__ data,
                const float* __restrict__ winE, const float* __restrict__ winD,
                const float* __restrict__ biasE, const float* __restrict__ biasD,
                const float* __restrict__ outW,
                const __bf16* __restrict__ wfragE, const __bf16* __restrict__ wfragD,
                __bf16* __restrict__ hbuf, float* __restrict__ ypart,
                unsigned* __restrict__ cnt) {
    extern __shared__ __align__(64) char smem[];
    __bf16* wl    = (__bf16*)(smem + WL_OFF);                    // 262144 B
    __bf16* hl    = (__bf16*)(smem + HL_OFF);                    //   4096 B
    float*  part  = (float*)(smem + PART_OFF);                   //   2048 B
    float*  gateL = (float*)(smem + PART_OFF + 2048);            //    256 B
    float*  winL  = gateL + 64;                                  //    256 B
    float*  biasL = winL + 64;                                   //    256 B
    float*  ybuf  = biasL + 64;                                  //     64 B
    float*  cSt   = ybuf + 16;                                   //     64 B

    const unsigned tid  = threadIdx.x;
    const unsigned g    = blockIdx.x;
    const unsigned lane = tid & 31u;
    const unsigned wv   = tid >> 5;                 // wave id 0..7
    const int hi8 = (lane & 16u) ? 8 : 0;
    unsigned round = 0;

    for (int phase = 0; phase < 2; ++phase) {
        const __bf16* wsrc = phase ? wfragD : wfragE;
        const float*  win  = phase ? winD : winE;     // W_ih[:,0] is contiguous
        const float*  bia  = phase ? biasD : biasE;

        // Pull this WG's 256KB weight slice into LDS via the async engine
        // (one-time per phase; ASYNCcnt-tracked, no VGPR round trip).
        {
            const unsigned long long src =
                (unsigned long long)(wsrc + (size_t)g * FRAG_ELEMS);
#pragma unroll
            for (int it = 0; it < 64; ++it) {
                const unsigned idx = tid + (unsigned)it * BLK;   // uint4 index
                asyncLds128(WL_OFF + idx * 16u, src + (unsigned long long)idx * 16u);
            }
            waitAsync0();
        }
        if (tid < 64) {
            const unsigned q = tid >> 4, n = tid & 15u;
            winL[tid]  = win[q * HDIM + g * SLICE + n];
            biasL[tid] = bia[q * HDIM + g * SLICE + n];
        }
        if (phase == 0 && tid < SLICE) cSt[tid] = 0.0f;   // c0 = 0; carries into decoder
        __syncthreads();
        gridBarrier(cnt, ++round);

        for (int t = 0; t < SEQ; ++t) {
            const float x = phase ? (float)t : data[t];

            // Stage full h (2048 bf16 = 4KB) into LDS: one async b128/thread.
            asyncLds128(HL_OFF + tid * 16u,
                        (unsigned long long)hbuf + (unsigned long long)tid * 16u);
            waitAsync0();
            __syncthreads();

            // Each wave: 8 K-chunks x 4 gate blocks of WMMA.
            v8f acc[4] = {v8f{0.f}, v8f{0.f}, v8f{0.f}, v8f{0.f}};
#pragma unroll
            for (int j = 0; j < 8; ++j) {
                const int kc = (int)wv + j * 8;
                // A fragment: h[kc*32 .. +32) replicated over all 16 M rows.
                v16bf a;
                const __bf16* hp = hl + kc * 32;
#pragma unroll
                for (int r = 0; r < 8; ++r) {
                    const int k = ((r & 4) ? 16 : 0) + hi8 + ((r & 3) << 1);
                    a[2 * r]     = hp[k];
                    a[2 * r + 1] = hp[k + 1];
                }
#pragma unroll
                for (int q = 0; q < 4; ++q) {
                    const v16bf b = *(const v16bf*)(wl +
                        (((unsigned)(q * KCHUNKS + kc) * 32u + lane) * 16u));
                    acc[q] = __builtin_amdgcn_wmma_f32_16x16x32_bf16(
                        false, a, false, b, (short)0, acc[q], false, false);
                }
            }
            // D is replicated over M: lanes 0..15, VGPR0 hold y_n (n = lane).
            if (lane < 16) {
#pragma unroll
                for (int q = 0; q < 4; ++q)
                    part[(wv * 4 + q) * 16 + lane] = acc[q][0];
            }
            __syncthreads();

            // Cross-wave K reduction + input/bias terms.
            if (tid < 64) {
                const unsigned q = tid >> 4, n = tid & 15u;
                float s = 0.f;
#pragma unroll
                for (int w = 0; w < 8; ++w) s += part[(w * 4 + q) * 16 + n];
                gateL[tid] = s + winL[tid] * x + biasL[tid];
            }
            __syncthreads();

            // Gate nonlinearities + cell/hidden update for our 16 units.
            if (tid < SLICE) {
                const float gi = sigf(gateL[tid]);
                const float gf = sigf(gateL[16 + tid]);
                const float gg = tanhf(gateL[32 + tid]);
                const float go = sigf(gateL[48 + tid]);
                const float cn = gf * cSt[tid] + gi * gg;
                cSt[tid] = cn;
                const float hn = go * tanhf(cn);
                hbuf[g * SLICE + tid] = (__bf16)hn;
                ybuf[tid] = outW[g * SLICE + tid] * hn;
            }
            __syncthreads();
            if (phase == 1 && tid == 0) {
                float s = 0.f;
#pragma unroll
                for (int n = 0; n < 16; ++n) s += ybuf[n];
                ypart[(size_t)t * NWG + g] = s;   // per-WG slot: no atomics
            }
            gridBarrier(cnt, ++round);   // publish h, sequence the recurrence
        }
    }
}

// ---------------------------------------------------------------------------
extern "C" void kernel_launch(void* const* d_in, const int* in_sizes, int n_in,
                              void* d_out, int out_size, void* d_ws, size_t ws_size,
                              hipStream_t stream) {
    const float* data = (const float*)d_in[0];
    const float* eWih = (const float*)d_in[1];
    const float* eWhh = (const float*)d_in[2];
    const float* ebih = (const float*)d_in[3];
    const float* ebhh = (const float*)d_in[4];
    const float* dWih = (const float*)d_in[5];
    const float* dWhh = (const float*)d_in[6];
    const float* dbih = (const float*)d_in[7];
    const float* dbhh = (const float*)d_in[8];
    const float* outW = (const float*)d_in[9];
    const float* outb = (const float*)d_in[10];
    float* y = (float*)d_out;

    // Workspace carve-up (256B aligned): ~66.5 MB total.
    char* ws = (char*)d_ws;
    size_t off = 0;
    auto carve = [&](size_t bytes) -> char* {
        char* p = ws + off;
        off = (off + bytes + 255) & ~(size_t)255;
        return p;
    };
    unsigned* cnt   = (unsigned*)carve(sizeof(unsigned));
    float*    biasE = (float*)carve((size_t)4 * HDIM * sizeof(float));
    float*    biasD = (float*)carve((size_t)4 * HDIM * sizeof(float));
    __bf16*   hbuf  = (__bf16*)carve((size_t)HDIM * sizeof(__bf16));
    float*    ypart = (float*)carve((size_t)SEQ * NWG * sizeof(float));
    __bf16*   wfragE = (__bf16*)carve((size_t)4 * HDIM * HDIM * sizeof(__bf16));
    __bf16*   wfragD = (__bf16*)carve((size_t)4 * HDIM * HDIM * sizeof(__bf16));

    init_kernel<<<(4 * HDIM + BLK - 1) / BLK, BLK, 0, stream>>>(
        ebih, ebhh, dbih, dbhh, biasE, biasD, hbuf, cnt);
    swizzle_whh<<<(4 * HDIM * HDIM) / BLK, BLK, 0, stream>>>(eWhh, wfragE);
    swizzle_whh<<<(4 * HDIM * HDIM) / BLK, BLK, 0, stream>>>(dWhh, wfragD);

    const size_t smem = 262144 + 4096 + 2048 + 256 + 256 + 256 + 64 + 64; // 269184 B
    lstm_persistent<<<NWG, BLK, smem, stream>>>(
        data, eWih, dWih, biasE, biasD, outW, wfragE, wfragD, hbuf, ypart, cnt);

    finalize_kernel<<<SEQ / BLK, BLK, 0, stream>>>(ypart, outb, y);
}
